// GraphNC_70738111365645
// MI455X (gfx1250) — matrix operand order repaired and checked by reading, not compiled, not run on hardware
//
#include <hip/hip_runtime.h>
#include <hip/hip_bf16.h>
#include <cstdint>

// ---------------------------------------------------------------------------
// GNN edge scorer: out[e] = sigmoid(W3 . relu(W2^T relu(W1^T [zi[src];zj[dst]])))
// One wave32 = one 32-edge pair-tile (two 16-row M tiles sharing every B tile:
// each ds_load_b128 pair of weights feeds two v_wmma_f32_16x16x32_f16).
// k-loops kept rolled to bound VGPR pressure (<256, no spills, no vgpr_msb).
// ---------------------------------------------------------------------------

typedef _Float16 v8h  __attribute__((ext_vector_type(8)));
typedef _Float16 v16h __attribute__((ext_vector_type(16)));
typedef float    v8f  __attribute__((ext_vector_type(8)));
typedef float    f4   __attribute__((ext_vector_type(4)));

#define WPB 8               // waves per block (256 threads)
#define SM_W1T   0          // 128*256 halfs = 65536 B  (W1 transposed: [col][k])
#define SM_W2T   65536      // 64*128 halfs  = 16384 B  (W2 transposed: [col][k])
#define SM_B1    81920      // 128 f32
#define SM_B2    82432      // 64 f32
#define SM_W3    82688      // 64 f32
#define SM_WAVE  82944      // per-wave scratch
#define H1_BYTES 8192       // h1: 32*128 f16
#define H2_BYTES 8704       // h2: 32*68 f32
#define WAVE_SCRATCH (H1_BYTES + H2_BYTES)          // 16896 B
#define SMEM_BYTES (SM_WAVE + WPB * WAVE_SCRATCH)   // 218112 B (< 320 KB WGP)

__device__ __forceinline__ v16h cat8(v8h lo, v8h hi) {
    return __builtin_shufflevector(lo, hi, 0,1,2,3,4,5,6,7,8,9,10,11,12,13,14,15);
}

// Load the two 8-float chunks of one A-layout k-step at p[0..7], p[16..23]
// and convert fp32 -> f16 into a 16-half A register.
__device__ __forceinline__ v16h load_a_f32(const float* p) {
    f4 f0 = *(const f4*)p,        f1 = *(const f4*)(p + 4);
    f4 f2 = *(const f4*)(p + 16), f3 = *(const f4*)(p + 20);
    v8f lo = __builtin_shufflevector(f0, f1, 0,1,2,3,4,5,6,7);
    v8f up = __builtin_shufflevector(f2, f3, 0,1,2,3,4,5,6,7);
    return cat8(__builtin_convertvector(lo, v8h),
                __builtin_convertvector(up, v8h));
}

__global__ void __launch_bounds__(WPB * 32)
edge_mlp_kernel(const float* __restrict__ zi, const float* __restrict__ zj,
                const int* __restrict__ src, const int* __restrict__ dst,
                const float* __restrict__ W1, const float* __restrict__ b1,
                const float* __restrict__ W2, const float* __restrict__ b2,
                const float* __restrict__ W3, const float* __restrict__ b3,
                float* __restrict__ out, int E, int nPairs) {
    extern __shared__ char smem[];
    _Float16* W1t = (_Float16*)(smem + SM_W1T);   // [128][256]
    _Float16* W2t = (_Float16*)(smem + SM_W2T);   // [64][128]
    float*    b1s = (float*)(smem + SM_B1);
    float*    b2s = (float*)(smem + SM_B2);
    float*    w3s = (float*)(smem + SM_W3);

    const int tid = threadIdx.x;
    const int nt  = blockDim.x;

    // ---- one-time per-block: stage transposed f16 weights + biases in LDS ----
    for (int idx = tid; idx < 256 * 128; idx += nt) {
        int k = idx >> 7, c = idx & 127;              // W1 is [256][128] row-major
        W1t[c * 256 + k] = (_Float16)W1[idx];
    }
    for (int idx = tid; idx < 128 * 64; idx += nt) {
        int k = idx >> 6, c = idx & 63;               // W2 is [128][64] row-major
        W2t[c * 128 + k] = (_Float16)W2[idx];
    }
    for (int idx = tid; idx < 128; idx += nt) b1s[idx] = b1[idx];
    for (int idx = tid; idx < 64; idx += nt) { b2s[idx] = b2[idx]; w3s[idx] = W3[idx]; }
    __syncthreads();

    const float b3v = b3[0];

    const int  l  = tid & 31;           // lane in wave
    const int  w  = tid >> 5;           // wave in block
    const int  m  = l & 15;             // M/N slot for this lane
    const bool hs = (l >= 16);          // high half of wave
    const int  kA = hs ? 8 : 0;         // A-layout K base (8-split pattern)
    const int  kB = hs ? 8 : 0;         // B-layout K base (8-split pattern)
    const int  Mlo = (hs ? 8 : 0);      // C/D row base for this lane half

    char* wbase = smem + SM_WAVE + w * WAVE_SCRATCH;
    _Float16* h1 = (_Float16*)wbase;                // [32][128] f16
    float*    h2 = (float*)(wbase + H1_BYTES);      // [32][68]  f32

    const int gw      = blockIdx.x * WPB + w;
    const int strideT = gridDim.x * WPB;

    for (int t = gw; t < nPairs; t += strideT) {
        const int e0 = t * 32;
        int ea = e0 + m;      if (ea >= E) ea = E - 1;   // tile0 edge (clamped)
        int eb = e0 + 16 + m; if (eb >= E) eb = E - 1;   // tile1 edge (clamped)
        const float* ri0 = zi + (size_t)src[ea] * 128;
        const float* rj0 = zj + (size_t)dst[ea] * 128;
        const float* ri1 = zi + (size_t)src[eb] * 128;
        const float* rj1 = zj + (size_t)dst[eb] * 128;

        int tn = t + strideT;
        if (tn < nPairs) {
            __builtin_prefetch(&src[tn * 32], 0, 0);
            __builtin_prefetch(&dst[tn * 32], 0, 0);
        }

        // ======= Layer 1: two [16,256]x[256,128] tiles sharing B loads =======
        v8f acc0[8], acc1[8];
        #pragma unroll
        for (int n = 0; n < 8; ++n) {
            acc0[n] = (v8f){0.f,0.f,0.f,0.f,0.f,0.f,0.f,0.f};
            acc1[n] = (v8f){0.f,0.f,0.f,0.f,0.f,0.f,0.f,0.f};
        }

        #pragma clang loop unroll(disable)
        for (int kk = 0; kk < 8; ++kk) {
            // Both 8-float chunks of this k-step sit on one side of the
            // zi/zj concat boundary (crossover exactly at kk==4).
            const float* base0 = (kk < 4) ? (ri0 + kk * 32) : (rj0 + (kk - 4) * 32);
            const float* base1 = (kk < 4) ? (ri1 + kk * 32) : (rj1 + (kk - 4) * 32);
            v16h a0 = load_a_f32(base0 + kA);
            v16h a1 = load_a_f32(base1 + kA);
            const int kb = kk * 32 + kB;
            #pragma unroll
            for (int n = 0; n < 8; ++n) {
                const _Float16* colp = W1t + (size_t)(n * 16 + m) * 256;
                v16h b = cat8(*(const v8h*)(colp + kb), *(const v8h*)(colp + kb + 16));
                acc0[n] = __builtin_amdgcn_wmma_f32_16x16x32_f16(
                    false, a0, false, b, (short)0, acc0[n], false, false);
                acc1[n] = __builtin_amdgcn_wmma_f32_16x16x32_f16(
                    false, a1, false, b, (short)0, acc1[n], false, false);
            }
        }

        // bias + relu -> f16 h1 in LDS ([32][128], tile1 in rows 16..31)
        #pragma unroll
        for (int n = 0; n < 8; ++n) {
            const int col = n * 16 + m;
            const float bias = b1s[col];
            #pragma unroll
            for (int r = 0; r < 8; ++r) {
                const int M = r + Mlo;
                h1[M * 128 + col]        = (_Float16)fmaxf(acc0[n][r] + bias, 0.f);
                h1[(16 + M) * 128 + col] = (_Float16)fmaxf(acc1[n][r] + bias, 0.f);
            }
        }

        // ======= Layer 2: two [16,128]x[128,64] tiles sharing B loads ========
        v8f c20[4], c21[4];
        #pragma unroll
        for (int n = 0; n < 4; ++n) {
            c20[n] = (v8f){0.f,0.f,0.f,0.f,0.f,0.f,0.f,0.f};
            c21[n] = (v8f){0.f,0.f,0.f,0.f,0.f,0.f,0.f,0.f};
        }

        #pragma clang loop unroll(disable)
        for (int kk = 0; kk < 4; ++kk) {
            const int k0 = kk * 32 + kA;
            const _Float16* ar0 = h1 + m * 128;
            const _Float16* ar1 = h1 + (16 + m) * 128;
            v16h a0 = cat8(*(const v8h*)(ar0 + k0), *(const v8h*)(ar0 + k0 + 16));
            v16h a1 = cat8(*(const v8h*)(ar1 + k0), *(const v8h*)(ar1 + k0 + 16));
            const int kb = kk * 32 + kB;
            #pragma unroll
            for (int n = 0; n < 4; ++n) {
                const _Float16* colp = W2t + (size_t)(n * 16 + m) * 128;
                v16h b = cat8(*(const v8h*)(colp + kb), *(const v8h*)(colp + kb + 16));
                c20[n] = __builtin_amdgcn_wmma_f32_16x16x32_f16(
                    false, a0, false, b, (short)0, c20[n], false, false);
                c21[n] = __builtin_amdgcn_wmma_f32_16x16x32_f16(
                    false, a1, false, b, (short)0, c21[n], false, false);
            }
        }

        // bias + relu -> f32 h2 in LDS ([32][68])
        #pragma unroll
        for (int n = 0; n < 4; ++n) {
            const int col = n * 16 + m;
            const float bias = b2s[col];
            #pragma unroll
            for (int r = 0; r < 8; ++r) {
                const int M = r + Mlo;
                h2[M * 68 + col]        = fmaxf(c20[n][r] + bias, 0.f);
                h2[(16 + M) * 68 + col] = fmaxf(c21[n][r] + bias, 0.f);
            }
        }

        // ======= Layer 3: one edge per lane, vectorized 64-dot + sigmoid =====
        {
            const int eo = e0 + l;
            if (eo < E) {
                const float* hr = h2 + l * 68;
                f4 s4 = (f4){0.f, 0.f, 0.f, 0.f};
                #pragma unroll
                for (int c = 0; c < 16; ++c)
                    s4 += (*(const f4*)(hr + 4 * c)) * (*(const f4*)(w3s + 4 * c));
                float s = b3v + s4.x + s4.y + s4.z + s4.w;
                out[eo] = 1.0f / (1.0f + __expf(-s));
            }
        }
    }
}

extern "C" void kernel_launch(void* const* d_in, const int* in_sizes, int n_in,
                              void* d_out, int out_size, void* d_ws, size_t ws_size,
                              hipStream_t stream) {
    const float* zi = (const float*)d_in[0];
    const float* zj = (const float*)d_in[1];
    const int*   src = (const int*)d_in[2];
    const int*   dst = (const int*)d_in[3];
    const float* W1 = (const float*)d_in[4];
    const float* b1 = (const float*)d_in[5];
    const float* W2 = (const float*)d_in[6];
    const float* b2 = (const float*)d_in[7];
    const float* W3 = (const float*)d_in[8];
    const float* b3 = (const float*)d_in[9];
    float* out = (float*)d_out;

    const int E = in_sizes[2];
    if (E <= 0) return;
    const int nPairs = (E + 31) / 32;

    int blocks = (nPairs + WPB - 1) / WPB;
    if (blocks > 1024) blocks = 1024;
    if (blocks < 1) blocks = 1;

    edge_mlp_kernel<<<blocks, WPB * 32, SMEM_BYTES, stream>>>(
        zi, zj, src, dst, W1, b1, W2, b2, W3, b3, out, E, nPairs);
}
